// MultiHeadContextualBiasedAttention_71373766525295
// MI455X (gfx1250) — matrix operand order repaired and checked by reading, not compiled, not run on hardware
//
#include <hip/hip_runtime.h>

// ---------------------------------------------------------------------------
// MI455X (gfx1250, wave32) fused cross-attention:
//   proj_gemm x3  -> bf16 Q[B,H,T1,64], K[B,H,T2,64], Vt[B,H,64,T2]
//   attn_kernel   -> flash attention with periodic ALiBi bias -> bf16 O
//   out_gemm      -> f32 out = O @ Wo^T + bo
// All matmuls on v_wmma_f32_16x16x32_bf16 (f32 accumulate).
// GEMMs are software-pipelined: next global tile is loaded into registers
// while WMMAs consume the current LDS tile. bf16 packing via v_perm_b32.
// ---------------------------------------------------------------------------

typedef __attribute__((ext_vector_type(16))) __bf16 v16bf;
typedef __attribute__((ext_vector_type(8)))  float  v8f;

#define NHEAD 16
#define DHEAD 64
#define T1LEN 2048
#define T2LEN 2048
#define DMODEL 1024
#define BATCH 2

#define BM 128
#define BN 128
#define BK 32

// round-half-up f32->bf16; pack two into one dword with a single v_perm_b32
static __device__ __forceinline__ unsigned pack2bf(float x, float y) {
    const unsigned ux = __float_as_uint(x) + 0x8000u;
    const unsigned uy = __float_as_uint(y) + 0x8000u;
    // dst bytes {0,1} = ux bytes {2,3}; dst bytes {2,3} = uy bytes {6,7 of {uy,ux}}
    return __builtin_amdgcn_perm(uy, ux, 0x07060302u);
}
static __device__ __forceinline__ __bf16 f2bf(float f) {
    const unsigned u = (__float_as_uint(f) + 0x8000u) >> 16;
    union { unsigned short u; __bf16 b; } cv; cv.u = (unsigned short)u;
    return cv.b;
}

static __device__ __forceinline__ v8f wmma_bf16(v16bf a, v16bf b, v8f c) {
    return __builtin_amdgcn_wmma_f32_16x16x32_bf16(false, a, false, b,
                                                   (short)0, c, false, false);
}

// ---------------------------------------------------------------------------
// Projection GEMM: out = A[M,K] @ W[N,K]^T, A,W f32, out bf16 in head layout.
// mode 0/1: out[((b*H+h)*2048 + t)*64 + d]   (Q / K)
// mode 2:   out[((b*H+h)*64 + d)*2048 + t]   (V transposed for P@V B-operand)
// M = 4096, N = K = 1024, T = 2048 fixed; all tile-exact (no bounds checks).
// ---------------------------------------------------------------------------
__global__ __launch_bounds__(256)
void proj_gemm(const float* __restrict__ A, const float* __restrict__ W,
               __bf16* __restrict__ out, int mode) {
    __shared__ __bf16 As[BM * BK];
    __shared__ __bf16 Bs[BN * BK];

    const int tid  = threadIdx.x;
    const int lane = tid & 31, wave = tid >> 5;
    const int hl   = lane >> 4, l16 = lane & 15;
    const int wm   = wave & 1,  wn  = wave >> 1;     // 2 x 4 wave grid
    const int m0   = blockIdx.y * BM, n0 = blockIdx.x * BN;

    v8f acc[4][2];
    for (int mi = 0; mi < 4; ++mi)
        for (int ni = 0; ni < 2; ++ni)
            for (int r = 0; r < 8; ++r) acc[mi][ni][r] = 0.0f;

    // per-thread staging map: 4 float4 chunks each for A and W tiles
    int sr[4], sc[4];
    for (int i = 0; i < 4; ++i) {
        const int idx = tid + i * 256;
        sr[i] = idx >> 3;            // row in tile (128 rows, 8 chunks/row)
        sc[i] = (idx & 7) << 2;      // column (float4 granularity)
    }

    float4 aReg[4], bReg[4];
    for (int i = 0; i < 4; ++i) {
        aReg[i] = *(const float4*)(A + (size_t)(m0 + sr[i]) * DMODEL + sc[i]);
        bReg[i] = *(const float4*)(W + (size_t)(n0 + sr[i]) * DMODEL + sc[i]);
    }

    for (int k0 = 0; k0 < DMODEL; k0 += BK) {
        // commit staged registers to LDS as packed bf16 (uint2 = 4 elems)
        for (int i = 0; i < 4; ++i) {
            uint2 pa; pa.x = pack2bf(aReg[i].x, aReg[i].y);
                      pa.y = pack2bf(aReg[i].z, aReg[i].w);
            *(uint2*)&As[sr[i] * BK + sc[i]] = pa;
            uint2 pb; pb.x = pack2bf(bReg[i].x, bReg[i].y);
                      pb.y = pack2bf(bReg[i].z, bReg[i].w);
            *(uint2*)&Bs[sr[i] * BK + sc[i]] = pb;
        }
        __syncthreads();

        // prefetch next k-tile into registers (overlaps with WMMAs below)
        if (k0 + BK < DMODEL) {
            const int kn = k0 + BK;
            for (int i = 0; i < 4; ++i) {
                aReg[i] = *(const float4*)(A + (size_t)(m0 + sr[i]) * DMODEL + kn + sc[i]);
                bReg[i] = *(const float4*)(W + (size_t)(n0 + sr[i]) * DMODEL + kn + sc[i]);
            }
        }

        // B fragments: lane = column n (l16), K chunks hl*8 and hl*8+16
        v16bf bfrag[2];
        for (int ni = 0; ni < 2; ++ni) {
            const __bf16* p = &Bs[(wn * 32 + ni * 16 + l16) * BK + hl * 8];
            ((float4*)&bfrag[ni])[0] = ((const float4*)p)[0];
            ((float4*)&bfrag[ni])[1] = *(const float4*)(p + 16);
        }
        // A fragments: lane = row m (l16), K chunks hl*8 and hl*8+16
        for (int mi = 0; mi < 4; ++mi) {
            const __bf16* p = &As[(wm * 64 + mi * 16 + l16) * BK + hl * 8];
            v16bf afrag;
            ((float4*)&afrag)[0] = *(const float4*)p;
            ((float4*)&afrag)[1] = *(const float4*)(p + 16);
            for (int ni = 0; ni < 2; ++ni)
                acc[mi][ni] = wmma_bf16(afrag, bfrag[ni], acc[mi][ni]);
        }
        __syncthreads();
    }

    // C layout: VGPR r -> row r + 8*hl, column l16. T = 2048 (shift/mask).
    for (int mi = 0; mi < 4; ++mi)
        for (int ni = 0; ni < 2; ++ni)
            for (int r = 0; r < 8; ++r) {
                const int m = m0 + wm * 64 + mi * 16 + r + 8 * hl;
                const int n = n0 + wn * 32 + ni * 16 + l16;
                const int b = m >> 11, t = m & 2047;
                const int h = n >> 6,  d = n & 63;
                const __bf16 v = f2bf(acc[mi][ni][r]);
                if (mode == 2)
                    out[((size_t)(b * NHEAD + h) * DHEAD + d) * T2LEN + t] = v;
                else
                    out[((size_t)(b * NHEAD + h) * T2LEN + t) * DHEAD + d] = v;
            }
}

// ---------------------------------------------------------------------------
// Flash attention with periodic ALiBi bias.
// Grid: (T1/64, B*H), block 128 (4 waves), wave w owns 16 Q rows.
// ---------------------------------------------------------------------------
__global__ __launch_bounds__(128)
void attn_kernel(const __bf16* __restrict__ Q, const __bf16* __restrict__ Km,
                 const __bf16* __restrict__ Vt, __bf16* __restrict__ O) {
    __shared__ __bf16 Ps[4][16 * 64];     // per-wave P staging (C->A relayout)

    const int tid  = threadIdx.x;
    const int lane = tid & 31, wave = tid >> 5;
    const int hl   = lane >> 4, l16 = lane & 15;
    const int bh   = blockIdx.y;          // b*NHEAD + h
    const int h    = bh & (NHEAD - 1);
    const int b    = bh >> 4;
    const int q0   = blockIdx.x * 64 + wave * 16;
    const float scale = 0.125f;           // 1/sqrt(64)
    const float hs = exp2f(-(float)(h + 1));

    // Q A-fragments for k=0..31 and 32..63 (loaded once)
    v16bf aq[2];
    {
        const __bf16* qp = Q + ((size_t)bh * T1LEN + q0 + l16) * DHEAD;
        for (int kk = 0; kk < 2; ++kk) {
            const __bf16* p = qp + kk * 32 + hl * 8;
            ((float4*)&aq[kk])[0] = *(const float4*)p;
            ((float4*)&aq[kk])[1] = *(const float4*)(p + 16);
        }
    }

    float rm[8], rl[8];
    v8f o[4];
    for (int r = 0; r < 8; ++r) { rm[r] = -1e30f; rl[r] = 0.0f; }
    for (int dt = 0; dt < 4; ++dt)
        for (int r = 0; r < 8; ++r) o[dt][r] = 0.0f;

    for (int jt = 0; jt < T2LEN; jt += 64) {
        // prefetch next K/V tiles (global_prefetch_b8); pattern is streaming
        if (jt + 64 < T2LEN) {
            __builtin_prefetch(Km + ((size_t)bh * T2LEN + jt + 64 + lane * 2) * DHEAD, 0, 1);
            __builtin_prefetch(Vt + ((size_t)bh * DHEAD + lane * 2) * T2LEN + jt + 64, 0, 1);
        }
        // ---- S = Q K^T for a 16x64 tile ----
        v8f s[4];
        for (int nt = 0; nt < 4; ++nt) {
            for (int r = 0; r < 8; ++r) s[nt][r] = 0.0f;
            const __bf16* kp = Km + ((size_t)bh * T2LEN + jt + nt * 16 + l16) * DHEAD;
            for (int kk = 0; kk < 2; ++kk) {
                v16bf bk;
                const __bf16* p = kp + kk * 32 + hl * 8;
                ((float4*)&bk)[0] = ((const float4*)p)[0];
                ((float4*)&bk)[1] = *(const float4*)(p + 16);
                s[nt] = wmma_bf16(aq[kk], bk, s[nt]);
            }
        }
        // ---- scale + periodic ALiBi bias + tile row max ----
        float pmax[8];
        for (int r = 0; r < 8; ++r) {
            const int i = q0 + r + 8 * hl;
            float mx = -1e30f;
            for (int nt = 0; nt < 4; ++nt) {
                const int j = jt + nt * 16 + l16;
                const int dist = (i > j) ? (i - j) : (j - i);
                const float v = s[nt][r] * scale - (float)(dist / 30) * hs;
                s[nt][r] = v;
                mx = fmaxf(mx, v);
            }
            for (int m = 1; m < 16; m <<= 1)
                mx = fmaxf(mx, __shfl_xor(mx, m, 32));
            pmax[r] = mx;
        }
        // ---- online softmax update ----
        for (int r = 0; r < 8; ++r) {
            const float nm = fmaxf(rm[r], pmax[r]);
            const float f  = __expf(rm[r] - nm);
            rm[r] = nm;
            float lsum = 0.0f;
            for (int nt = 0; nt < 4; ++nt) {
                const float p = __expf(s[nt][r] - nm);
                s[nt][r] = p;
                lsum += p;
            }
            for (int m = 1; m < 16; m <<= 1)
                lsum += __shfl_xor(lsum, m, 32);
            rl[r] = rl[r] * f + lsum;
            for (int dt = 0; dt < 4; ++dt) o[dt][r] *= f;
        }
        // ---- stage P (C layout) to LDS row-major 16x64 bf16 ----
        __bf16* ps = &Ps[wave][0];
        for (int nt = 0; nt < 4; ++nt)
            for (int r = 0; r < 8; ++r)
                ps[(r + 8 * hl) * 64 + nt * 16 + l16] = f2bf(s[nt][r]);
        // ---- reload P as A fragments (same-wave DS ops are in order) ----
        v16bf ap[2];
        const __bf16* pp = ps + l16 * 64;
        for (int kk = 0; kk < 2; ++kk) {
            const __bf16* p = pp + kk * 32 + hl * 8;
            ((float4*)&ap[kk])[0] = *(const float4*)p;
            ((float4*)&ap[kk])[1] = *(const float4*)(p + 16);
        }
        // ---- O += P @ V  (Vt: [bh][d][t2], contiguous in t2 = K dim) ----
        for (int dt = 0; dt < 4; ++dt) {
            const __bf16* vp = Vt + ((size_t)bh * DHEAD + dt * 16 + l16) * T2LEN + jt;
            for (int kk = 0; kk < 2; ++kk) {
                v16bf bv;
                const __bf16* p = vp + kk * 32 + hl * 8;
                ((float4*)&bv)[0] = ((const float4*)p)[0];
                ((float4*)&bv)[1] = *(const float4*)(p + 16);
                o[dt] = wmma_bf16(ap[kk], bv, o[dt]);
            }
        }
    }

    // ---- normalize and store O as bf16 [B, T1, H*64] ----
    for (int dt = 0; dt < 4; ++dt)
        for (int r = 0; r < 8; ++r) {
            const int i = q0 + r + 8 * hl;
            const float v = o[dt][r] / rl[r];
            O[((size_t)(b * T1LEN + i)) * DMODEL + h * DHEAD + dt * 16 + l16] = f2bf(v);
        }
}

// ---------------------------------------------------------------------------
// Output projection: out = O[M,K](bf16) @ Wo[N,K]^T + bo, f32 result.
// Software-pipelined like proj_gemm.
// ---------------------------------------------------------------------------
__global__ __launch_bounds__(256)
void out_gemm(const __bf16* __restrict__ A, const float* __restrict__ W,
              const float* __restrict__ bias, float* __restrict__ out) {
    __shared__ __bf16 As[BM * BK];
    __shared__ __bf16 Bs[BN * BK];

    const int tid  = threadIdx.x;
    const int lane = tid & 31, wave = tid >> 5;
    const int hl   = lane >> 4, l16 = lane & 15;
    const int wm   = wave & 1,  wn  = wave >> 1;
    const int m0   = blockIdx.y * BM, n0 = blockIdx.x * BN;

    v8f acc[4][2];
    for (int mi = 0; mi < 4; ++mi)
        for (int ni = 0; ni < 2; ++ni)
            for (int r = 0; r < 8; ++r) acc[mi][ni][r] = 0.0f;

    // A staging: 2 x float4 (=8 bf16) chunks; W staging: 4 x float4 chunks
    int ar[2], ac[2], wr[4], wc[4];
    for (int i = 0; i < 2; ++i) {
        const int idx = tid + i * 256;
        ar[i] = idx >> 2;            // 4 chunks of 8 bf16 per 32-wide row
        ac[i] = (idx & 3) << 3;
    }
    for (int i = 0; i < 4; ++i) {
        const int idx = tid + i * 256;
        wr[i] = idx >> 3;
        wc[i] = (idx & 7) << 2;
    }

    float4 aReg[2], bReg[4];
    for (int i = 0; i < 2; ++i)
        aReg[i] = *(const float4*)(A + (size_t)(m0 + ar[i]) * DMODEL + ac[i]);
    for (int i = 0; i < 4; ++i)
        bReg[i] = *(const float4*)(W + (size_t)(n0 + wr[i]) * DMODEL + wc[i]);

    for (int k0 = 0; k0 < DMODEL; k0 += BK) {
        for (int i = 0; i < 2; ++i)
            *(float4*)&As[ar[i] * BK + ac[i]] = aReg[i];
        for (int i = 0; i < 4; ++i) {
            uint2 pb; pb.x = pack2bf(bReg[i].x, bReg[i].y);
                      pb.y = pack2bf(bReg[i].z, bReg[i].w);
            *(uint2*)&Bs[wr[i] * BK + wc[i]] = pb;
        }
        __syncthreads();

        if (k0 + BK < DMODEL) {
            const int kn = k0 + BK;
            for (int i = 0; i < 2; ++i)
                aReg[i] = *(const float4*)(A + (size_t)(m0 + ar[i]) * DMODEL + kn + ac[i]);
            for (int i = 0; i < 4; ++i)
                bReg[i] = *(const float4*)(W + (size_t)(n0 + wr[i]) * DMODEL + kn + wc[i]);
        }

        v16bf bfrag[2];
        for (int ni = 0; ni < 2; ++ni) {
            const __bf16* p = &Bs[(wn * 32 + ni * 16 + l16) * BK + hl * 8];
            ((float4*)&bfrag[ni])[0] = ((const float4*)p)[0];
            ((float4*)&bfrag[ni])[1] = *(const float4*)(p + 16);
        }
        for (int mi = 0; mi < 4; ++mi) {
            const __bf16* p = &As[(wm * 64 + mi * 16 + l16) * BK + hl * 8];
            v16bf afrag;
            ((float4*)&afrag)[0] = *(const float4*)p;
            ((float4*)&afrag)[1] = *(const float4*)(p + 16);
            for (int ni = 0; ni < 2; ++ni)
                acc[mi][ni] = wmma_bf16(afrag, bfrag[ni], acc[mi][ni]);
        }
        __syncthreads();
    }

    for (int mi = 0; mi < 4; ++mi)
        for (int ni = 0; ni < 2; ++ni)
            for (int r = 0; r < 8; ++r) {
                const int m = m0 + wm * 64 + mi * 16 + r + 8 * hl;
                const int n = n0 + wn * 32 + ni * 16 + l16;
                out[(size_t)m * DMODEL + n] = acc[mi][ni][r] + bias[n];
            }
}

// ---------------------------------------------------------------------------
extern "C" void kernel_launch(void* const* d_in, const int* in_sizes, int n_in,
                              void* d_out, int out_size, void* d_ws, size_t ws_size,
                              hipStream_t stream) {
    (void)in_sizes; (void)n_in; (void)out_size; (void)ws_size;
    const float* x   = (const float*)d_in[0];
    const float* ctx = (const float*)d_in[1];
    const float* Wq  = (const float*)d_in[2];
    const float* Wk  = (const float*)d_in[3];
    const float* Wv  = (const float*)d_in[4];
    const float* Wo  = (const float*)d_in[5];
    const float* bo  = (const float*)d_in[6];
    float* out = (float*)d_out;

    const size_t QKV_ELEMS = (size_t)BATCH * NHEAD * T2LEN * DHEAD;  // 4M bf16
    __bf16* q  = (__bf16*)d_ws;
    __bf16* k  = q  + QKV_ELEMS;
    __bf16* vt = k  + QKV_ELEMS;
    __bf16* ob = vt + QKV_ELEMS;   // [B, T1, 1024] bf16

    const dim3 gg(DMODEL / BN, (BATCH * T1LEN) / BM);   // (8, 32)
    proj_gemm<<<gg, 256, 0, stream>>>(x,   Wq, q,  0);
    proj_gemm<<<gg, 256, 0, stream>>>(ctx, Wk, k,  1);
    proj_gemm<<<gg, 256, 0, stream>>>(ctx, Wv, vt, 2);
    attn_kernel<<<dim3(T1LEN / 64, BATCH * NHEAD), 128, 0, stream>>>(q, k, vt, ob);
    out_gemm<<<gg, 256, 0, stream>>>(ob, Wo, bo, out);
}
